// T2S_trans_skip_85899345920261
// MI455X (gfx1250) — matrix operand order repaired and checked by reading, not compiled, not run on hardware
//
#include <hip/hip_runtime.h>
#include <hip/hip_bf16.h>

// ---------------------------------------------------------------------------
// T2S spatio-temporal transformer forward for MI455X (gfx1250).
// All heavy einsums -> one batched WMMA GEMM kernel:
//   v_wmma_f32_16x16x32_f16, fp32->f16 staged via LDS, fp32 accum.
// AI ~300 FLOP/byte >> HBM balance point at 23.3 TB/s => matrix-core bound.
// Interior tiles: async global->LDS (ASYNCcnt) double-buffered staging when
// the toolchain exposes the builtin; otherwise batched vector loads.
// ---------------------------------------------------------------------------

typedef __attribute__((ext_vector_type(16))) _Float16 v16h;
typedef __attribute__((ext_vector_type(8)))  _Float16 v8h;
typedef __attribute__((ext_vector_type(8)))  float    v8f;

#define TM 64
#define TN 64
#define TK 32
#define KVCH 64

#ifndef __has_builtin
#define __has_builtin(x) 0
#endif
#if defined(__HIP_DEVICE_COMPILE__) && __has_builtin(__builtin_amdgcn_global_load_async_to_lds_b32)
#define HAVE_ASYNC 1
#else
#define HAVE_ASYNC 0
#endif

#if HAVE_ASYNC
typedef __attribute__((address_space(1))) int as1_int;
typedef __attribute__((address_space(3))) int as3_int;
__device__ __forceinline__ void async_cp_f32(const float* g, float* lds) {
    __builtin_amdgcn_global_load_async_to_lds_b32(
        (as1_int*)(unsigned long long)g,
        (as3_int*)(unsigned int)(unsigned long long)lds,
        0, 0);
}
__device__ __forceinline__ void wait_async0() {
#if __has_builtin(__builtin_amdgcn_s_wait_asynccnt)
    __builtin_amdgcn_s_wait_asynccnt(0);
#else
    asm volatile("s_wait_asynccnt 0" ::: "memory");
#endif
}
#endif

__device__ __forceinline__ float apply_act(float v, int act) {
    if (act == 1) return 0.5f * v * (1.0f + erff(v * 0.70710678118654752440f)); // exact gelu
    if (act == 2) return (v > 0.0f) ? (v + 1.0f) : __expf(v);                   // elu(x)+1
    if (act == 3) return fmaxf(v, 0.0f);
    return v;
}

// Fragment build + 2x WMMA for one wave (16x32 sub-tile of the 64x64 C tile).
__device__ __forceinline__ void frag_mma(const _Float16* lA, const _Float16* lB,
                                         int lane, int m0, int n0, v8f& c0, v8f& c1)
{
    // A frag (16x32): lanes 0-15 K 0-15, lanes 16-31 K 16-31
    const int arow = m0 + (lane & 15);
    const int kb   = (lane >> 4) ? 16 : 0;
    const v8h* pa  = (const v8h*)(&lA[arow * TK + kb]);
    v8h alo = pa[0];
    v8h ahi = pa[1];   // +8 halves -> K = kb+8 .. kb+15
    v16h af;
#pragma unroll
    for (int j = 0; j < 8; ++j) { af[j] = alo[j]; af[j + 8] = ahi[j]; }
    // B frags: lanes 0-15 K 0-15, lanes 16-31 K 16-31 (contiguous in [n][k] tile)
    const int kb2 = (lane >> 4) ? 16 : 0;
    const v16h b0 = *(const v16h*)(&lB[(n0 +      (lane & 15)) * TK + kb2]);
    const v16h b1 = *(const v16h*)(&lB[(n0 + 16 + (lane & 15)) * TK + kb2]);
    c0 = __builtin_amdgcn_wmma_f32_16x16x32_f16(false, af, false, b0, (short)0, c0, false, false);
    c1 = __builtin_amdgcn_wmma_f32_16x16x32_f16(false, af, false, b1, (short)0, c1, false, false);
}

// Batched GEMM: C[z] = act( opA(A[z]) * B[sel(z)] + bias[sel(z)] ) (+ C_old)
// A: MxK (row-major, or K-major if transA), B: KxN row-major, C: MxN row-major.
__global__ __launch_bounds__(256) void gemm_wmma_f16(
    const float* __restrict__ A, long long lda, long long sA, int transA,
    const float* __restrict__ Bm, long long ldb, long long sB, int mB,
    const float* __restrict__ bias, long long sBias, int mBias,
    float* __restrict__ C, long long ldc, long long sC,
    int M, int K, int Nc, int act, int addC)
{
    __shared__ alignas(64) _Float16 lA[TM * TK];   // [m][k] f16
    __shared__ alignas(64) _Float16 lB[TN * TK];   // [n][k] f16 (transposed)
#if HAVE_ASYNC
    __shared__ alignas(64) float lA32[2][TM * TK]; // fp32 async staging (ping-pong)
    __shared__ alignas(64) float lB32[2][TN * TK]; // [k][n] row-major
#endif

    const int z = blockIdx.z;
    A  += (long long)z * sA;
    Bm += (long long)(mB == 0 ? z : (z % mB)) * sB;
    const float* bptr = bias ? bias + (long long)(mBias == 0 ? z : (z % mBias)) * sBias
                             : (const float*)nullptr;
    C  += (long long)z * sC;

    const int mblk = blockIdx.y * TM;
    const int nblk = blockIdx.x * TN;
    const int tid  = threadIdx.x;
    const int lane = tid & 31;
    const int wid  = tid >> 5;           // 8 waves (wave32)
    const int m0   = (wid & 3) * 16;
    const int n0   = (wid >> 2) * 32;

    v8f c0 = {}, c1 = {};

#if HAVE_ASYNC
    const bool fullTile = (mblk + TM <= M) && (nblk + TN <= Nc) && ((K & (TK - 1)) == 0);
    if (fullTile) {
        auto issue_tile = [&](int kkb, int buf) {
#pragma unroll
            for (int e = tid; e < TM * TK; e += 256) {
                int r = e >> 5, c = e & 31;
                const float* g = transA ? &A[(long long)(kkb + c) * lda + (mblk + r)]
                                        : &A[(long long)(mblk + r) * lda + (kkb + c)];
                async_cp_f32(g, &lA32[buf][e]);
            }
#pragma unroll
            for (int e = tid; e < TN * TK; e += 256) {
                int kl = e >> 6, nl = e & 63;   // coalesced along n
                async_cp_f32(&Bm[(long long)(kkb + kl) * ldb + (nblk + nl)], &lB32[buf][e]);
            }
        };
        const int nsteps = K / TK;
        issue_tile(0, 0);
        for (int s = 0; s < nsteps; ++s) {
            wait_async0();
            __syncthreads();                       // staging buffer visible to all waves
            const int cur = s & 1;
#pragma unroll
            for (int e = tid; e < TM * TK; e += 256) lA[e] = (_Float16)lA32[cur][e];
#pragma unroll
            for (int e = tid; e < TN * TK; e += 256) {
                int kl = e >> 6, nl = e & 63;
                lB[nl * TK + kl] = (_Float16)lB32[cur][e];
            }
            __syncthreads();                       // f16 tiles ready
            if (s + 1 < nsteps) issue_tile((s + 1) * TK, (s + 1) & 1); // overlap DMA w/ WMMA
            frag_mma(lA, lB, lane, m0, n0, c0, c1);
            __syncthreads();
        }
    } else
#endif
    {
        for (int kk = 0; kk < K; kk += TK) {
            const bool inter = (mblk + TM <= M) && (nblk + TN <= Nc) && (kk + TK <= K);
            if (kk + TK < K) {
                __builtin_prefetch(transA ? &A[(long long)(kk + TK) * lda + mblk]
                                          : &A[(long long)mblk * lda + (kk + TK)], 0, 3);
                __builtin_prefetch(&Bm[(long long)(kk + TK) * ldb + nblk], 0, 3);
            }
            if (inter) {
                // fast path: straight-line loads, one wait, no predicates
#pragma unroll
                for (int e = tid; e < TM * TK; e += 256) {
                    int r = e >> 5, c = e & 31;
                    float v = transA ? A[(long long)(kk + c) * lda + (mblk + r)]
                                     : A[(long long)(mblk + r) * lda + (kk + c)];
                    lA[e] = (_Float16)v;
                }
#pragma unroll
                for (int e = tid; e < TN * TK; e += 256) {
                    int kl = e >> 6, nl = e & 63;
                    lB[nl * TK + kl] = (_Float16)Bm[(long long)(kk + kl) * ldb + (nblk + nl)];
                }
            } else {
                for (int e = tid; e < TM * TK; e += 256) {
                    int r = e >> 5, c = e & 31;
                    int gm = mblk + r, gk = kk + c;
                    float v = 0.0f;
                    if (gm < M && gk < K)
                        v = transA ? A[(long long)gk * lda + gm] : A[(long long)gm * lda + gk];
                    lA[e] = (_Float16)v;
                }
                for (int e = tid; e < TN * TK; e += 256) {
                    int kl = e >> 6, nl = e & 63;
                    int gk = kk + kl, gn = nblk + nl;
                    float v = 0.0f;
                    if (gk < K && gn < Nc) v = Bm[(long long)gk * ldb + gn];
                    lB[nl * TK + kl] = (_Float16)v;
                }
            }
            __syncthreads();
            frag_mma(lA, lB, lane, m0, n0, c0, c1);
            __syncthreads();
        }
    }

    // ---- epilogue: bias + activation (+accumulate), bounds-checked store
    const int nc0   = nblk + n0 + (lane & 15);
    const int nc1   = nc0 + 16;
    const int mbase = mblk + m0 + ((lane >> 4) ? 8 : 0);
#pragma unroll
    for (int j = 0; j < 8; ++j) {
        int gm = mbase + j;
        if (gm >= M) continue;
        if (nc0 < Nc) {
            float v = c0[j];
            if (bptr) v += bptr[nc0];
            v = apply_act(v, act);
            if (addC) v += C[(long long)gm * ldc + nc0];
            C[(long long)gm * ldc + nc0] = v;
        }
        if (nc1 < Nc) {
            float v = c1[j];
            if (bptr) v += bptr[nc1];
            v = apply_act(v, act);
            if (addC) v += C[(long long)gm * ldc + nc1];
            C[(long long)gm * ldc + nc1] = v;
        }
    }
}

// ---------------------------------------------------------------------------
// Linear attention small pieces (E=16, H=8, D=128).
// Pass 1: per-(b,h,chunk) partial kv/ksum (deterministic fixed chunking).
__global__ __launch_bounds__(256) void kv_partial_kernel(
    const float* __restrict__ kp, const float* __restrict__ v,
    float* __restrict__ pkv, float* __restrict__ pks, int T)
{
    const int bh = blockIdx.x / KVCH, ch = blockIdx.x % KVCH;
    const int b = bh >> 3, h = bh & 7;
    const int e = threadIdx.x >> 4, dd = threadIdx.x & 15;
    const int t0 = (int)(((long long)T * ch) / KVCH);
    const int t1 = (int)(((long long)T * (ch + 1)) / KVCH);
    const float* kpb = kp + (long long)b * T * 128 + h * 16;
    const float* vb  = v  + (long long)b * T * 128 + h * 16;
    float acc = 0.0f, acc2 = 0.0f;
    for (int t = t0; t < t1; ++t) {
        float kk = kpb[(long long)t * 128 + e];
        float vv = vb [(long long)t * 128 + dd];
        acc += kk * vv;
        if (dd == 0) acc2 += kk;
    }
    pkv[((long long)bh * KVCH + ch) * 256 + threadIdx.x] = acc;
    if (dd == 0) pks[((long long)bh * KVCH + ch) * 16 + e] = acc2;
}

// Pass 2: deterministic finalize over the KVCH partials.
__global__ __launch_bounds__(256) void kv_finalize_kernel(
    const float* __restrict__ pkv, const float* __restrict__ pks,
    float* __restrict__ kv, float* __restrict__ ksum)
{
    const int bh = blockIdx.x;
    const int e = threadIdx.x >> 4, dd = threadIdx.x & 15;
    float acc = 0.0f, acc2 = 0.0f;
    for (int ch = 0; ch < KVCH; ++ch) {
        acc += pkv[((long long)bh * KVCH + ch) * 256 + threadIdx.x];
        if (dd == 0) acc2 += pks[((long long)bh * KVCH + ch) * 16 + e];
    }
    kv[(long long)bh * 256 + threadIdx.x] = acc;
    if (dd == 0) ksum[(long long)bh * 16 + e] = acc2;
}

// out[r,h*16+d] = (qp[r,h,:] . kv[b,h,:,d]) / (qp[r,h,:] . ksum[b,h,:] + 1e-6)
__global__ __launch_bounds__(128) void lin_attn_out_kernel(
    const float* __restrict__ qp, const float* __restrict__ kv,
    const float* __restrict__ ksum, float* __restrict__ out, int S)
{
    const long long r = blockIdx.x;
    const int b = (int)(r / S);
    const int col = threadIdx.x, h = col >> 4, d = col & 15;
    __shared__ float qs[128];
    qs[col] = qp[r * 128 + col];
    __syncthreads();
    const float* ks  = ksum + ((long long)b * 8 + h) * 16;
    const float* kvb = kv   + (((long long)b * 8 + h) * 16) * 16;
    float zz = 0.0f, o = 0.0f;
#pragma unroll
    for (int e = 0; e < 16; ++e) {
        float q = qs[h * 16 + e];
        zz += q * ks[e];
        o  += q * kvb[e * 16 + d];
    }
    out[r * 128 + col] = o / (zz + 1e-6f);
}

// LayerNorm over D=128 (optional residual add / relu-on-input).
__global__ __launch_bounds__(128) void ln_kernel(
    const float* __restrict__ x, const float* __restrict__ res,
    const float* __restrict__ g, const float* __restrict__ bb,
    float* __restrict__ out, int reluIn)
{
    const long long r = blockIdx.x;
    const int t = threadIdx.x;
    __shared__ float sh[128];
    float u = x[r * 128 + t];
    if (res) u += res[r * 128 + t];
    if (reluIn) u = fmaxf(u, 0.0f);
    sh[t] = u; __syncthreads();
    for (int s = 64; s > 0; s >>= 1) { if (t < s) sh[t] += sh[t + s]; __syncthreads(); }
    float mu = sh[0] * (1.0f / 128.0f);
    __syncthreads();
    float xc = u - mu;
    sh[t] = xc * xc; __syncthreads();
    for (int s = 64; s > 0; s >>= 1) { if (t < s) sh[t] += sh[t + s]; __syncthreads(); }
    float var = sh[0] * (1.0f / 128.0f);
    out[r * 128 + t] = xc * rsqrtf(var + 1e-5f) * g[t] + bb[t];
}

// data [BL,N,D] -> dtm [BL,D,N]
__global__ void transpose_nd_kernel(const float* __restrict__ in, float* __restrict__ outp,
                                    long long total, int N, int D)
{
    long long i = (long long)blockIdx.x * blockDim.x + threadIdx.x;
    if (i >= total) return;
    long long bl = i / ((long long)D * N);
    long long rem = i - bl * (long long)D * N;
    int c = (int)(rem / N), n = (int)(rem % N);
    outp[i] = in[(bl * N + n) * D + c];
}

__global__ void copy_kernel(const float* __restrict__ a, float* __restrict__ b, long long n) {
    long long i = (long long)blockIdx.x * blockDim.x + threadIdx.x;
    if (i < n) b[i] = a[i];
}
__global__ void zero_kernel(float* __restrict__ a, long long n) {
    long long i = (long long)blockIdx.x * blockDim.x + threadIdx.x;
    if (i < n) a[i] = 0.0f;
}

// ---------------------------------------------------------------------------
static void gemm(hipStream_t s,
                 const float* A, long long lda, long long sA, int transA,
                 const float* Bm, long long ldb, long long sB, int mB,
                 const float* bias, long long sBias, int mBias,
                 float* C, long long ldc, long long sC,
                 int M, int K, int Nc, int batch, int act, int addC)
{
    dim3 grid((Nc + TN - 1) / TN, (M + TM - 1) / TM, batch);
    gemm_wmma_f16<<<grid, 256, 0, s>>>(A, lda, sA, transA, Bm, ldb, sB, mB,
                                       bias, sBias, mBias, C, ldc, sC,
                                       M, K, Nc, act, addC);
}

extern "C" void kernel_launch(void* const* d_in, const int* in_sizes, int n_in,
                              void* d_out, int out_size, void* d_ws, size_t ws_size,
                              hipStream_t stream)
{
    (void)in_sizes; (void)n_in; (void)out_size; (void)ws_size;
    const int B = 8, L = 24, LP = 48, N = 325, D = 128, NL = 3, DFF = 512;
    const int BL = B * L;                       // 192
    const int S = L * N, T = LP * N;            // 7800, 15600
    const long long BS = (long long)B * S;      // 62400 rows
    const long long BT = (long long)B * T;      // 124800 rows
    const long long DN = (long long)D * N;      // 41600
    const long long NN = (long long)N * N;      // 105625

    const float* x_in    = (const float*)d_in[0];
    const float* memory  = (const float*)d_in[1];
    const float* data    = (const float*)d_in[2];
    const float* support = (const float*)d_in[3];
    const float* Wq = (const float*)d_in[4];  const float* bq = (const float*)d_in[5];
    const float* Wk = (const float*)d_in[6];  const float* bk = (const float*)d_in[7];
    const float* Wv = (const float*)d_in[8];  const float* bv = (const float*)d_in[9];
    const float* Wo = (const float*)d_in[10]; const float* bo = (const float*)d_in[11];
    const float* W1 = (const float*)d_in[12]; const float* b1 = (const float*)d_in[13];
    const float* W2 = (const float*)d_in[14]; const float* b2 = (const float*)d_in[15];
    const float* W3 = (const float*)d_in[16]; const float* b3 = (const float*)d_in[17];
    const float* lng = (const float*)d_in[18]; const float* lnb = (const float*)d_in[19];
    const float* gcnW = (const float*)d_in[20]; const float* gcnb = (const float*)d_in[21];
    const float* skW = (const float*)d_in[22]; const float* skb = (const float*)d_in[23];
    const float* outg = (const float*)d_in[24]; const float* outb = (const float*)d_in[25];
    float* out = (float*)d_out;

    // ---- workspace carve-out (floats) ----
    float* ws = (float*)d_ws;
    long long o = 0;
    float* f_xcur = ws + o; o += BL * DN;          // [BL,N,D]
    float* f_dtm  = ws + o; o += BL * DN;          // [BL,D,N]
    float* f_s1   = ws + o; o += BL * DN;
    float* f_s2   = ws + o; o += BL * DN;
    float* f_a1   = ws + o; o += BL * DN;
    float* f_a2   = ws + o; o += BL * DN;
    float* f_q    = ws + o; o += BS * D;
    float* f_k    = ws + o; o += BT * D;
    float* f_v    = ws + o; o += BT * D;
    float* f_ao   = ws + o; o += BS * D;
    float* f_y2   = ws + o; o += BS * D;           // proj / FFN output (time-shared)
    float* f_y1   = ws + o; o += BS * DFF;
    float* f_As   = ws + o; o += (long long)BL * NN;
    float* f_dts  = ws + o; o += BS * D;
    float* f_skip = ws + o; o += BS * D;
    float* f_kv   = ws + o; o += (long long)B * 8 * 16 * 16;
    float* f_ksum = ws + o; o += (long long)B * 8 * 16;
    float* f_pkv  = ws + o; o += (long long)B * 8 * KVCH * 256;
    float* f_pks  = ws + o; o += (long long)B * 8 * KVCH * 16;

    const long long XEL = BL * DN;  // 7,987,200
    copy_kernel<<<(int)((XEL + 255) / 256), 256, 0, stream>>>(x_in, f_xcur, XEL);
    zero_kernel<<<(int)((XEL + 255) / 256), 256, 0, stream>>>(f_skip, XEL);
    transpose_nd_kernel<<<(int)((XEL + 255) / 256), 256, 0, stream>>>(data, f_dtm, XEL, N, D);

    // static diffusion: s1 = dtm @ support, s2 = s1 @ support   (batched over BL)
    gemm(stream, f_dtm, N, DN, 0, support, N, 0, 1, nullptr, 0, 1,
         f_s1, N, DN, D, N, N, BL, 0, 0);
    gemm(stream, f_s1, N, DN, 0, support, N, 0, 1, nullptr, 0, 1,
         f_s2, N, DN, D, N, N, BL, 0, 0);

    for (int i = 0; i < NL; ++i) {
        for (int s = 0; s < 2; ++s) {   // s=0 self, s=1 cross
            const long long wOff = (long long)(i * 2 + s) * D * D;
            const long long bOff = (long long)(i * 2 + s) * D;
            const float* kvsrc = (s == 0) ? f_xcur : memory;
            const long long Trows = (s == 0) ? BS : BT;
            const int Ttok = (s == 0) ? S : T;
            // projections (elu+1 feature map fused into q,k epilogues)
            gemm(stream, f_xcur, D, 0, 0, Wq + wOff, D, 0, 1, bq + bOff, 0, 1,
                 f_q, D, 0, (int)BS, D, D, 1, 2, 0);
            gemm(stream, kvsrc, D, 0, 0, Wk + wOff, D, 0, 1, bk + bOff, 0, 1,
                 f_k, D, 0, (int)Trows, D, D, 1, 2, 0);
            gemm(stream, kvsrc, D, 0, 0, Wv + wOff, D, 0, 1, bv + bOff, 0, 1,
                 f_v, D, 0, (int)Trows, D, D, 1, 0, 0);
            kv_partial_kernel<<<B * 8 * KVCH, 256, 0, stream>>>(f_k, f_v, f_pkv, f_pks, Ttok);
            kv_finalize_kernel<<<B * 8, 256, 0, stream>>>(f_pkv, f_pks, f_kv, f_ksum);
            lin_attn_out_kernel<<<(int)BS, 128, 0, stream>>>(f_q, f_kv, f_ksum, f_ao, S);
            gemm(stream, f_ao, D, 0, 0, Wo + wOff, D, 0, 1, bo + bOff, 0, 1,
                 f_y2, D, 0, (int)BS, D, D, 1, 0, 0);
            ln_kernel<<<(int)BS, 128, 0, stream>>>(f_xcur, f_y2,
                lng + (long long)(i * 3 + s) * D, lnb + (long long)(i * 3 + s) * D, f_xcur, 0);
        }

        // FFN: y1 = gelu(x@W1+b1); y2 = y1@W2+b2; As = y2@W3+b3
        gemm(stream, f_xcur, D, 0, 0, W1 + (long long)i * D * DFF, DFF, 0, 1,
             b1 + (long long)i * DFF, 0, 1, f_y1, DFF, 0, (int)BS, D, DFF, 1, 1, 0);
        gemm(stream, f_y1, DFF, 0, 0, W2 + (long long)i * DFF * D, D, 0, 1,
             b2 + (long long)i * D, 0, 1, f_y2, D, 0, (int)BS, DFF, D, 1, 0, 0);
        gemm(stream, f_y2, D, 0, 0, W3 + (long long)i * D * N, N, 0, 1,
             b3 + (long long)i * N, 0, 1, f_As, N, 0, (int)BS, D, N, 1, 0, 0);

        // adaptive diffusion: a1 = dtm @ As[bl]; a2 = a1 @ As[bl]
        gemm(stream, f_dtm, N, DN, 0, f_As, N, NN, 0, nullptr, 0, 1,
             f_a1, N, DN, D, N, N, BL, 0, 0);
        gemm(stream, f_a1, N, DN, 0, f_As, N, NN, 0, nullptr, 0, 1,
             f_a2, N, DN, D, N, N, BL, 0, 0);

        // GCN: dts[bl] = sum_j chunk_j^T @ gcnW[i,l,j-block]  (+ gcnb[i,l] on j==0)
        const float* chunks[5] = { f_dtm, f_s1, f_s2, f_a1, f_a2 };
        for (int j = 0; j < 5; ++j) {
            const float* Wc = gcnW + ((long long)i * L * 5 * D + (long long)j * D) * D;
            gemm(stream, chunks[j], N, DN, 1 /*transA*/,
                 Wc, D, (long long)5 * D * D, L,
                 (j == 0) ? gcnb + (long long)i * L * D : nullptr, D, L,
                 f_dts, D, DN, N, D, D, BL, 0, (j > 0) ? 1 : 0);
        }

        // skip += dts @ skW[i] + skb[i]
        gemm(stream, f_dts, D, 0, 0, skW + (long long)i * D * D, D, 0, 1,
             skb + (long long)i * D, 0, 1, f_skip, D, 0, (int)BS, D, D, 1, 0, 1);

        // x = LN(x + y2)
        ln_kernel<<<(int)BS, 128, 0, stream>>>(f_xcur, f_y2,
            lng + (long long)(i * 3 + 2) * D, lnb + (long long)(i * 3 + 2) * D, f_xcur, 0);
    }

    // out = LN(relu(skip))
    ln_kernel<<<(int)BS, 128, 0, stream>>>(f_skip, nullptr, outg, outb, out, 1);
}
